// LigandGNNV2_60567628808810
// MI455X (gfx1250) — matrix operand order repaired and checked by reading, not compiled, not run on hardware
//
#include <hip/hip_runtime.h>
#include <hip/hip_bf16.h>
#include <math.h>

#define HDIM 128
#define HDIM2 256
#define IN_FEAT 1070

typedef __attribute__((ext_vector_type(16))) __bf16 v16bf;
typedef __attribute__((ext_vector_type(8)))  float  v8f;

union Frag16 { v16bf v; uint4 q[2]; };

static __device__ __forceinline__ unsigned f2bf_raw(float f) {
  unsigned u = __float_as_uint(f);
  u += 0x7FFFu + ((u >> 16) & 1u);   // round-to-nearest-even truncation
  return u >> 16;
}
static __device__ __forceinline__ unsigned short f2bf(float f) {
  return (unsigned short)f2bf_raw(f);
}
static __device__ __forceinline__ unsigned pack2bf(float lo, float hi) {
  return f2bf_raw(lo) | (f2bf_raw(hi) << 16);
}

static __device__ __forceinline__ void atomicMaxF(float* addr, float v) {
  // monotone bit-pattern trick; buffer initialized to -inf (0xFF800000)
  if (v >= 0.0f) atomicMax((int*)addr, __float_as_int(v));
  else           atomicMin((unsigned int*)addr, __float_as_uint(v));
}

// ---------------------------------------------------------------------------
// Tiled bf16 WMMA GEMM: C[M,Nc] = A[M,K] * W[K,Nc] + bias (+ residual)
// block = 256 threads (8 waves), tile 64x128, BK=32; 4 WMMA per wave per k-step
// ---------------------------------------------------------------------------
__global__ __launch_bounds__(256)
void wmma_gemm_bf16(const float* __restrict__ A, const float* __restrict__ W,
                    const float* __restrict__ bias, const float* __restrict__ residual,
                    float* __restrict__ C, int M, int K, int Nc)
{
  constexpr int BM = 64, BN = 128, BK = 32;
  constexpr int LDA = 40, LDB = 40;      // padded halves; rows stay 16-B aligned (80 B)
  __shared__ unsigned short As[BM * LDA] __attribute__((aligned(16)));
  __shared__ unsigned short Bs[BN * LDB] __attribute__((aligned(16)));

  const int tid  = threadIdx.x;
  const int lane = tid & 31;
  const int wave = tid >> 5;
  const int half = lane >> 4;            // 0 | 1
  const int l16  = lane & 15;

  const int row0 = blockIdx.y * BM;
  const int col0 = blockIdx.x * BN;
  const int mrow = (wave >> 1) * 16;     // 0,16,32,48
  const int chalf = (wave & 1) * 64;     // 0 | 64 (each wave owns 64 cols)

  v8f acc[4] = {v8f{}, v8f{}, v8f{}, v8f{}};

  const bool edgeM = (row0 + BM > M);
  const bool edgeN = (col0 + BN > Nc);

  const int ksteps = (K + BK - 1) / BK;
  for (int ks = 0; ks < ksteps; ++ks) {
    const int kb = ks * BK;
    const bool tailK = (kb + BK > K);
    __syncthreads();

    // ---- stage A tile (row-major bf16), 64 x 32 ----
    if (!edgeM && !tailK) {
      #pragma unroll
      for (int i = 0; i < 4; ++i) {
        int v = i * 256 + tid;               // 1024 float2 units
        int r = v >> 4, kd = (v & 15) * 2;   // 16 float2 per row
        float2 a2 = *(const float2*)(A + (size_t)(row0 + r) * K + kb + kd);
        *(unsigned*)(&As[r * LDA + kd]) = pack2bf(a2.x, a2.y);
      }
      // prefetch next A k-tile (global_prefetch_b8)
      if (kb + BK < K) {
        int pr = row0 + (tid & 63);
        int pk = kb + BK + ((tid >> 6) << 3);
        if (pk < K) __builtin_prefetch(A + (size_t)pr * K + pk, 0, 3);
      }
    } else {
      #pragma unroll
      for (int i = 0; i < 8; ++i) {
        int idx = i * 256 + tid;
        int r = idx >> 5, k = idx & 31;
        int gr = row0 + r, gk = kb + k;
        float v = (gr < M && gk < K) ? A[(size_t)gr * K + gk] : 0.0f;
        As[r * LDA + k] = f2bf(v);
      }
    }

    // ---- stage B tile transposed: Bs[c][k], 128 x 32 ----
    if (!edgeN && !tailK) {
      #pragma unroll
      for (int i = 0; i < 4; ++i) {
        int v = i * 256 + tid;               // 1024 units of (c, 4 consecutive k)
        int c = v & 127, kq = (v >> 7) * 4;
        const float* wp = W + (size_t)(kb + kq) * Nc + col0 + c;
        float w0 = wp[0];
        float w1 = wp[(size_t)Nc];
        float w2 = wp[(size_t)2 * Nc];
        float w3 = wp[(size_t)3 * Nc];
        *(uint2*)(&Bs[c * LDB + kq]) = make_uint2(pack2bf(w0, w1), pack2bf(w2, w3));
      }
    } else {
      #pragma unroll
      for (int i = 0; i < 16; ++i) {
        int idx = i * 256 + tid;
        int k = idx >> 7, c = idx & 127;
        int gk = kb + k, gc = col0 + c;
        float v = (gk < K && gc < Nc) ? W[(size_t)gk * Nc + gc] : 0.0f;
        Bs[c * LDB + k] = f2bf(v);
      }
    }
    __syncthreads();

    // A fragment: lanes 0-15 hold K in [0..7]u[16..23], lanes 16-31 [8..15]u[24..31]
    Frag16 a;
    const unsigned short* ap = &As[(mrow + l16) * LDA + 8 * half];
    a.q[0] = *(const uint4*)(ap);
    a.q[1] = *(const uint4*)(ap + 16);

    #pragma unroll
    for (int t = 0; t < 4; ++t) {
      // B fragment: column l16 of tile t, K in [8*half..+7]u[16+8*half..+7] (matches A)
      Frag16 b;
      const unsigned short* bp = &Bs[(chalf + t * 16 + l16) * LDB + 8 * half];
      b.q[0] = *(const uint4*)(bp);
      b.q[1] = *(const uint4*)(bp + 16);
      acc[t] = __builtin_amdgcn_wmma_f32_16x16x32_bf16(false, a.v, false, b.v,
                                                       (short)0, acc[t], false, false);
    }
  }

  // epilogue: acc VGPR r -> M = r + 8*half, N = l16
  const int orow = row0 + mrow + 8 * half;
  #pragma unroll
  for (int r = 0; r < 8; ++r) {
    int gm = orow + r;
    if (gm >= M) continue;
    #pragma unroll
    for (int t = 0; t < 4; ++t) {
      int gc = col0 + chalf + t * 16 + l16;
      if (gc >= Nc) continue;
      size_t o = (size_t)gm * Nc + gc;
      float v = acc[t][r] + bias[gc];
      if (residual) v += residual[o];
      C[o] = v;
    }
  }
}

// ---------------------------------------------------------------------------
// Edge pass 1: per-(dst,feature) max of logits (32 lanes/edge, 4 feats/lane)
// ---------------------------------------------------------------------------
__global__ __launch_bounds__(256)
void edge_max_kernel(const float* __restrict__ hin, const int* __restrict__ src,
                     const int* __restrict__ dst, const float* __restrict__ tptr,
                     int layer, float* __restrict__ nmax, int E)
{
  int tid = blockIdx.x * blockDim.x + threadIdx.x;
  int e = tid >> 5;
  if (e >= E) return;
  int f0 = (tid & 31) << 2;
  float t = tptr[layer];
  int s = src[e], d = dst[e];
  float4 hv = *(const float4*)(hin + (size_t)s * HDIM + f0);
  float* np = nmax + (size_t)d * HDIM + f0;
  atomicMaxF(np + 0, (fmaxf(hv.x, 0.f) + 1e-7f) * t);
  atomicMaxF(np + 1, (fmaxf(hv.y, 0.f) + 1e-7f) * t);
  atomicMaxF(np + 2, (fmaxf(hv.z, 0.f) + 1e-7f) * t);
  atomicMaxF(np + 3, (fmaxf(hv.w, 0.f) + 1e-7f) * t);
}

// ---------------------------------------------------------------------------
// Edge pass 2: num += msg*exp(logit-max), den += exp(logit-max)
// ---------------------------------------------------------------------------
__global__ __launch_bounds__(256)
void edge_sum_kernel(const float* __restrict__ hin, const int* __restrict__ src,
                     const int* __restrict__ dst, const float* __restrict__ tptr,
                     int layer, const float* __restrict__ nmax,
                     float* __restrict__ num, float* __restrict__ den, int E)
{
  int tid = blockIdx.x * blockDim.x + threadIdx.x;
  int e = tid >> 5;
  if (e >= E) return;
  int f0 = (tid & 31) << 2;
  float t = tptr[layer];
  int s = src[e], d = dst[e];
  float4 hv = *(const float4*)(hin + (size_t)s * HDIM + f0);
  float4 mv = *(const float4*)(nmax + (size_t)d * HDIM + f0);
  float m0 = fmaxf(hv.x, 0.f) + 1e-7f, m1 = fmaxf(hv.y, 0.f) + 1e-7f;
  float m2 = fmaxf(hv.z, 0.f) + 1e-7f, m3 = fmaxf(hv.w, 0.f) + 1e-7f;
  float e0 = __expf(m0 * t - mv.x), e1 = __expf(m1 * t - mv.y);
  float e2 = __expf(m2 * t - mv.z), e3 = __expf(m3 * t - mv.w);
  float* dp = den + (size_t)d * HDIM + f0;
  float* np = num + (size_t)d * HDIM + f0;
  atomicAdd(dp + 0, e0); atomicAdd(dp + 1, e1);
  atomicAdd(dp + 2, e2); atomicAdd(dp + 3, e3);
  atomicAdd(np + 0, m0 * e0); atomicAdd(np + 1, m1 * e1);
  atomicAdd(np + 2, m2 * e2); atomicAdd(np + 3, m3 * e3);
}

__global__ __launch_bounds__(256)
void init_accum_kernel(float* __restrict__ nmax, float* __restrict__ num,
                       float* __restrict__ den, int total4)
{
  int i = blockIdx.x * blockDim.x + threadIdx.x;
  if (i >= total4) return;
  float ninf = __uint_as_float(0xFF800000u);  // -inf
  ((float4*)nmax)[i] = make_float4(ninf, ninf, ninf, ninf);
  ((float4*)num )[i] = make_float4(0.f, 0.f, 0.f, 0.f);
  ((float4*)den )[i] = make_float4(0.f, 0.f, 0.f, 0.f);
}

// xin = num/(den+1e-16) + hin   (root residual inside GENConv)
__global__ __launch_bounds__(256)
void finalize_agg_kernel(const float* __restrict__ num, const float* __restrict__ den,
                         const float* __restrict__ hin, float* __restrict__ xin, int total4)
{
  int i = blockIdx.x * blockDim.x + threadIdx.x;
  if (i >= total4) return;
  float4 n = ((const float4*)num)[i];
  float4 d = ((const float4*)den)[i];
  float4 h = ((const float4*)hin)[i];
  float4 o;
  o.x = n.x / (d.x + 1e-16f) + h.x;
  o.y = n.y / (d.y + 1e-16f) + h.y;
  o.z = n.z / (d.z + 1e-16f) + h.z;
  o.w = n.w / (d.w + 1e-16f) + h.w;
  ((float4*)xin)[i] = o;
}

// ---------------------------------------------------------------------------
// LayerNorm (+optional ReLU), one wave32 per row, in-place safe
// ---------------------------------------------------------------------------
__global__ __launch_bounds__(256)
void ln_act_kernel(const float* __restrict__ in, const float* __restrict__ w,
                   const float* __restrict__ b, float* __restrict__ out,
                   int rows, int D, int doRelu)
{
  int wid  = (blockIdx.x * blockDim.x + threadIdx.x) >> 5;
  int lane = threadIdx.x & 31;
  if (wid >= rows) return;
  const float* xr = in + (size_t)wid * D;
  float s = 0.f, s2 = 0.f;
  for (int j = lane; j < D; j += 32) { float v = xr[j]; s += v; s2 += v * v; }
  for (int o = 16; o > 0; o >>= 1) { s += __shfl_xor(s, o, 32); s2 += __shfl_xor(s2, o, 32); }
  float inv = 1.0f / (float)D;
  float mu = s * inv;
  float var = s2 * inv - mu * mu;
  float rs = rsqrtf(var + 1e-5f);
  float* orow = out + (size_t)wid * D;
  for (int j = lane; j < D; j += 32) {
    float v = (xr[j] - mu) * rs * w[j] + b[j];
    orow[j] = doRelu ? fmaxf(v, 0.f) : v;
  }
}

// final head: out[n] = relu(LN(h; ln_w[0], ln_b[0])) . lin_W + lin_b
__global__ __launch_bounds__(256)
void head_kernel(const float* __restrict__ hbuf, const float* __restrict__ lnw,
                 const float* __restrict__ lnb, const float* __restrict__ linW,
                 const float* __restrict__ linb, float* __restrict__ out, int rows)
{
  int wid  = (blockIdx.x * blockDim.x + threadIdx.x) >> 5;
  int lane = threadIdx.x & 31;
  if (wid >= rows) return;
  const float* xr = hbuf + (size_t)wid * HDIM;
  float s = 0.f, s2 = 0.f;
  #pragma unroll
  for (int j = lane; j < HDIM; j += 32) { float v = xr[j]; s += v; s2 += v * v; }
  for (int o = 16; o > 0; o >>= 1) { s += __shfl_xor(s, o, 32); s2 += __shfl_xor(s2, o, 32); }
  float mu = s * (1.0f / HDIM);
  float var = s2 * (1.0f / HDIM) - mu * mu;
  float rs = rsqrtf(var + 1e-5f);
  float acc = 0.f;
  #pragma unroll
  for (int j = lane; j < HDIM; j += 32) {
    float v = (xr[j] - mu) * rs * lnw[j] + lnb[j];
    acc += fmaxf(v, 0.f) * linW[j];
  }
  for (int o = 16; o > 0; o >>= 1) acc += __shfl_xor(acc, o, 32);
  if (lane == 0) out[wid] = acc + linb[0];
}

// ---------------------------------------------------------------------------
extern "C" void kernel_launch(void* const* d_in, const int* in_sizes, int n_in,
                              void* d_out, int out_size, void* d_ws, size_t ws_size,
                              hipStream_t stream)
{
  const float* x     = (const float*)d_in[0];
  const int*   ei    = (const int*)  d_in[1];
  const float* enc_W = (const float*)d_in[2];
  const float* enc_b = (const float*)d_in[3];
  const float* tptr  = (const float*)d_in[4];
  const float* w1    = (const float*)d_in[5];
  const float* b1    = (const float*)d_in[6];
  const float* mln_w = (const float*)d_in[7];
  const float* mln_b = (const float*)d_in[8];
  const float* w2    = (const float*)d_in[9];
  const float* b2    = (const float*)d_in[10];
  const float* ln_w  = (const float*)d_in[11];
  const float* ln_b  = (const float*)d_in[12];
  const float* lin_W = (const float*)d_in[13];
  const float* lin_b = (const float*)d_in[14];

  const int N = in_sizes[0] / IN_FEAT;
  const int E = in_sizes[1] / 2;
  const int L = in_sizes[4];
  const int* src  = ei;
  const int* dstp = ei + E;

  // workspace layout (fp32): h | hn | nmax(=xin) | num | den | z1(2H)
  const size_t NH = (size_t)N * HDIM;
  float* ws   = (float*)d_ws;
  float* h    = ws;
  float* hn   = h + NH;
  float* nmax = hn + NH;
  float* num  = nmax + NH;
  float* den  = num + NH;
  float* z1   = den + NH;          // N * 2H
  float* xin  = nmax;              // reuse once max pass is consumed

  dim3 blk(256);
  const int edge_blocks = (E * 32 + 255) / 256;
  const int nh4   = (int)(NH / 4);
  const int initg = (nh4 + 255) / 256;
  const int rowsg = (N + 7) / 8;   // 8 wave32 rows per block

  // encoder: h = x @ enc_W + enc_b
  {
    dim3 g((HDIM + 127) / 128, (N + 63) / 64);
    wmma_gemm_bf16<<<g, blk, 0, stream>>>(x, enc_W, enc_b, nullptr, h, N, IN_FEAT, HDIM);
  }

  for (int l = 0; l < L; ++l) {
    const float* hin;
    if (l == 0) {
      hin = h;
    } else {
      ln_act_kernel<<<rowsg, blk, 0, stream>>>(h, ln_w + (size_t)l * HDIM,
                                               ln_b + (size_t)l * HDIM, hn, N, HDIM, 1);
      hin = hn;
    }

    init_accum_kernel<<<initg, blk, 0, stream>>>(nmax, num, den, nh4);
    edge_max_kernel<<<edge_blocks, blk, 0, stream>>>(hin, src, dstp, tptr, l, nmax, E);
    edge_sum_kernel<<<edge_blocks, blk, 0, stream>>>(hin, src, dstp, tptr, l, nmax, num, den, E);
    finalize_agg_kernel<<<initg, blk, 0, stream>>>(num, den, hin, xin, nh4);

    // z1 = xin @ w1[l] + b1[l]   (H -> 2H)
    {
      dim3 g((HDIM2 + 127) / 128, (N + 63) / 64);
      wmma_gemm_bf16<<<g, blk, 0, stream>>>(xin, w1 + (size_t)l * HDIM * HDIM2,
                                            b1 + (size_t)l * HDIM2, nullptr, z1,
                                            N, HDIM, HDIM2);
    }
    // z1 = relu(LN(z1)) in place
    ln_act_kernel<<<rowsg, blk, 0, stream>>>(z1, mln_w + (size_t)l * HDIM2,
                                             mln_b + (size_t)l * HDIM2, z1, N, HDIM2, 1);
    // h = (l==0 ? 0 : h) + z1 @ w2[l] + b2[l]   (2H -> H, fused residual)
    {
      dim3 g((HDIM + 127) / 128, (N + 63) / 64);
      wmma_gemm_bf16<<<g, blk, 0, stream>>>(z1, w2 + (size_t)l * HDIM2 * HDIM,
                                            b2 + (size_t)l * HDIM,
                                            (l == 0) ? nullptr : h, h,
                                            N, HDIM2, HDIM);
    }
  }

  head_kernel<<<rowsg, blk, 0, stream>>>(h, ln_w, ln_b, lin_W, lin_b, (float*)d_out, N);
}
